// PointNet2_glr_80917183857080
// MI455X (gfx1250) — matrix operand order repaired and checked by reading, not compiled, not run on hardware
//
#include <hip/hip_runtime.h>
#include <hip/hip_bf16.h>

// PointNet++ SA pipeline for MI455X (gfx1250, wave32, WMMA).
// GEMMs run as v_wmma_f32_16x16x32_f16 over pre-swizzled fragment buffers.
// KT (K-tiles) and OB (output-tile blocking) are template parameters:
//  - A-fragments (weights) register-cached across the n-tile loop
//  - each B-fragment load feeds OB independent WMMA accumulator chains
//    (halves B traffic, doubles register-level FLOP/byte for OB=2).

typedef _Float16 v16h __attribute__((ext_vector_type(16)));
typedef float    v8f  __attribute__((ext_vector_type(8)));

#define NBATCH 32
#define NPTS   2048

// ---------------------------------------------------------------- utilities
__device__ inline unsigned ballot32(bool p) {
#if __has_builtin(__builtin_amdgcn_ballot_w32)
  return __builtin_amdgcn_ballot_w32(p);
#else
  return (unsigned)__ballot((int)p);
#endif
}
__device__ inline unsigned lanemask_lt32(int lane) {
  return (lane == 0) ? 0u : (0xffffffffu >> (32 - lane));
}

__device__ inline v8f wmma16(v16h a, v16h b, v8f c) {
  return __builtin_amdgcn_wmma_f32_16x16x32_f16(
      /*neg_a=*/false, a, /*neg_b=*/false, b,
      /*c_mod=*/(short)0, c, /*reuse_a=*/false, /*reuse_b=*/false);
}

// Load OB*KT A-fragments (weights) for output tiles ot0..ot0+OB-1.
template <int KT, int OB>
__device__ inline void load_afrag(const v16h* __restrict__ A, int ot0, int lane,
                                  v16h* __restrict__ a) {
#pragma unroll
  for (int ob = 0; ob < OB; ++ob)
#pragma unroll
    for (int kt = 0; kt < KT; ++kt)
      a[ob * KT + kt] = A[((long)(ot0 + ob) * KT + kt) * 32 + lane];
}

// OB 16x16 D tiles from cached A and a single streamed B fragment per kt.
template <int KT, int OB>
__device__ inline void wmma_tiles(const v16h* __restrict__ a,
                                  const v16h* __restrict__ Bm,
                                  long nt, int lane, v8f* __restrict__ acc) {
#pragma unroll
  for (int ob = 0; ob < OB; ++ob) acc[ob] = (v8f){};
#pragma unroll
  for (int kt = 0; kt < KT; ++kt) {
    v16h b = Bm[(nt * KT + kt) * 32 + lane];
#pragma unroll
    for (int ob = 0; ob < OB; ++ob)
      acc[ob] = wmma16(a[ob * KT + kt], b, acc[ob]);
  }
}

// ---------------------------------------------------------------- weights
// Pack W[O x C] (f32) into WMMA A-fragments (16x32 f16 tiles).
// A layout (contiguous k order, same as B packing): lane m=lane&15 holds row M=m; element e holds
// K = kt*32 + ((lane>>4)*16) + e.
__global__ void pack_weights_kernel(const float* __restrict__ W,
                                    _Float16* __restrict__ WA,
                                    int O, int C, int OT, int KT) {
  int t = blockIdx.x * blockDim.x + threadIdx.x;
  int total = OT * KT * 32;
  if (t >= total) return;
  int lane = t & 31;
  int kt   = (t >> 5) % KT;
  int ot   = t / (32 * KT);
  int o    = ot * 16 + (lane & 15);
  int kb   = (lane >> 4) * 16;
  v16h v;
  for (int e = 0; e < 16; ++e) {
    int K = kt * 32 + kb + e;
    float val = (o < O && K < C) ? W[(long)o * C + K] : 0.f;
    v[e] = (_Float16)val;
  }
  reinterpret_cast<v16h*>(WA)[t] = v;
}

// ---------------------------------------------------------------- FPS
// One 256-thread block per batch. dist[] lives in LDS (8KB of 320KB/WGP).
__global__ void fps_kernel(const float* __restrict__ pts,
                           int bstride, int nstride, int cstride,
                           int Nsrc, int M,
                           int* __restrict__ idx_out,
                           float* __restrict__ xyz_out) {
  int b = blockIdx.x, tid = threadIdx.x;
  __shared__ float dist[NPTS];
  __shared__ float rv[256];
  __shared__ int   ri[256];
  __shared__ float spx, spy, spz;
  const float* P = pts + (long)b * bstride;

  for (int i = tid; i < Nsrc; i += 256) dist[i] = 1e10f;
  if (tid == 0) {
    idx_out[b * M] = 0;
    spx = P[0]; spy = P[cstride]; spz = P[2 * cstride];
    xyz_out[(long)(b * M) * 3 + 0] = spx;
    xyz_out[(long)(b * M) * 3 + 1] = spy;
    xyz_out[(long)(b * M) * 3 + 2] = spz;
  }
  __syncthreads();

  for (int m = 1; m < M; ++m) {
    float bx = spx, by = spy, bz = spz;
    float bv = -1.f; int bi = 0;
    for (int i = tid; i < Nsrc; i += 256) {
      long base = (long)i * nstride;
      float dx = P[base] - bx;
      float dy = P[base + cstride] - by;
      float dz = P[base + 2 * cstride] - bz;
      float d = dx * dx + dy * dy + dz * dz;
      float dd = fminf(dist[i], d);
      dist[i] = dd;
      if (dd > bv) { bv = dd; bi = i; }
    }
    rv[tid] = bv; ri[tid] = bi;
    __syncthreads();
    for (int s = 128; s > 0; s >>= 1) {
      if (tid < s) {
        if (rv[tid + s] > rv[tid] ||
            (rv[tid + s] == rv[tid] && ri[tid + s] < ri[tid])) {
          rv[tid] = rv[tid + s]; ri[tid] = ri[tid + s];
        }
      }
      __syncthreads();
    }
    if (tid == 0) {
      int nxt = ri[0];
      idx_out[b * M + m] = nxt;
      long base = (long)nxt * nstride;
      spx = P[base]; spy = P[base + cstride]; spz = P[base + 2 * cstride];
      xyz_out[(long)(b * M + m) * 3 + 0] = spx;
      xyz_out[(long)(b * M + m) * 3 + 1] = spy;
      xyz_out[(long)(b * M + m) * 3 + 2] = spz;
    }
    __syncthreads();
  }
}

// ---------------------------------------------------------------- grouping
// Stage-1 ball query + feature build. One wave32 per query.
// Writes 6 channels [rel(3), gx(3)] into B-fragment layout, Kpad=32.
__global__ void group1_kernel(const float* __restrict__ pc,
                              const float* __restrict__ xyz1,
                              _Float16* __restrict__ frag) {
  int lane = threadIdx.x & 31;
  int wv   = threadIdx.x >> 5;
  int q    = blockIdx.x * 2 + wv;                 // [0, 32*512)
  if (q >= NBATCH * 512) return;
  int b = q >> 9;
  __shared__ int list[2][48];
  const float* P = pc + (long)b * 3 * NPTS;       // (3, 2048)
  float cx = xyz1[(long)q * 3], cy = xyz1[(long)q * 3 + 1], cz = xyz1[(long)q * 3 + 2];
  const float r2 = 0.23f * 0.23f;
  int cnt = 0;
  for (int base = 0; base < NPTS && cnt < 48; base += 32) {
    int i = base + lane;
    float dx = P[i] - cx, dy = P[NPTS + i] - cy, dz = P[2 * NPTS + i] - cz;
    bool in = (dx * dx + dy * dy + dz * dz) < r2;
    unsigned msk = ballot32(in);
    int rank = __popc(msk & lanemask_lt32(lane));
    if (in && (cnt + rank) < 48) list[wv][cnt + rank] = i;
    cnt += __popc(msk);
  }
  if (cnt > 48) cnt = 48;
  asm volatile("s_wait_dscnt 0" ::: "memory");    // wave-level LDS fence
  int first = (cnt > 0) ? list[wv][0] : 0;

  v16h z = {};
  for (int k = lane; k < 48; k += 32) {
    int nb = (k < cnt) ? list[wv][k] : first;
    float gx = P[nb], gy = P[NPTS + nb], gz = P[2 * NPTS + nb];
    v16h v = {};
    v[0] = (_Float16)(gx - cx); v[1] = (_Float16)(gy - cy); v[2] = (_Float16)(gz - cz);
    v[3] = (_Float16)gx;        v[4] = (_Float16)gy;        v[5] = (_Float16)gz;
    long n  = (long)q * 48 + k;
    long nt = n >> 4;
    int  nl = (int)(n & 15);
    reinterpret_cast<v16h*>(frag)[nt * 32 + nl]      = v;   // K 0..15 in lanes 0..15
    reinterpret_cast<v16h*>(frag)[nt * 32 + nl + 16] = z;   // zero pad (c>=16)
  }
}

// Stage-2 ball query over xyz1 (512 pts) + gather of f1 (128 ch).
// Channels: rel(3), gx(3), f1(128) -> Kpad = 160 (KT=5).
__global__ void group2_kernel(const float* __restrict__ xyz1,
                              const float* __restrict__ xyz2,
                              const _Float16* __restrict__ f1p,   // (b,s1,128)
                              _Float16* __restrict__ frag) {
  int lane = threadIdx.x & 31;
  int wv   = threadIdx.x >> 5;
  int q    = blockIdx.x * 2 + wv;                 // [0, 32*128)
  if (q >= NBATCH * 128) return;
  int b = q >> 7;
  __shared__ int list[2][64];
  const float* S = xyz1 + (long)b * 512 * 3;      // (512, 3)
  float cx = xyz2[(long)q * 3], cy = xyz2[(long)q * 3 + 1], cz = xyz2[(long)q * 3 + 2];
  const float r2 = 0.32f * 0.32f;
  int cnt = 0;
  for (int base = 0; base < 512 && cnt < 64; base += 32) {
    int i = base + lane;
    float dx = S[i * 3] - cx, dy = S[i * 3 + 1] - cy, dz = S[i * 3 + 2] - cz;
    bool in = (dx * dx + dy * dy + dz * dz) < r2;
    unsigned msk = ballot32(in);
    int rank = __popc(msk & lanemask_lt32(lane));
    if (in && (cnt + rank) < 64) list[wv][cnt + rank] = i;
    cnt += __popc(msk);
  }
  if (cnt > 64) cnt = 64;
  asm volatile("s_wait_dscnt 0" ::: "memory");
  int first = (cnt > 0) ? list[wv][0] : 0;

  for (int k = lane; k < 64; k += 32) {
    int nb = (k < cnt) ? list[wv][k] : first;
    float gx = S[nb * 3], gy = S[nb * 3 + 1], gz = S[nb * 3 + 2];
    float rel[6] = { gx - cx, gy - cy, gz - cz, gx, gy, gz };
    const _Float16* f1r = f1p + ((long)b * 512 + nb) * 128;
    long n    = (long)q * 64 + k;
    long base = (n >> 4) * (5 * 32) + (n & 15);
    for (int kt = 0; kt < 5; ++kt)
      for (int hi = 0; hi < 2; ++hi) {
        v16h v;
        for (int e = 0; e < 16; ++e) {
          int c = kt * 32 + hi * 16 + e;
          float val = (c < 6) ? rel[c]
                    : (c < 134) ? (float)f1r[c - 6]
                    : 0.f;
          v[e] = (_Float16)val;
        }
        reinterpret_cast<v16h*>(frag)[base + kt * 32 + hi * 16] = v;
      }
  }
}

// Stage-3 input: concat(xyz2(3), f2(512)) -> Kpad=544 (KT=17), one thread/n.
__global__ void gather3_kernel(const float* __restrict__ xyz2,
                               const _Float16* __restrict__ f2p,  // (b,s2,512)
                               _Float16* __restrict__ frag) {
  int n = blockIdx.x * blockDim.x + threadIdx.x;  // b*128 + s2
  if (n >= NBATCH * 128) return;
  float x0 = xyz2[(long)n * 3], x1 = xyz2[(long)n * 3 + 1], x2 = xyz2[(long)n * 3 + 2];
  const _Float16* fr = f2p + (long)n * 512;
  long nt = n >> 4;
  int  nl = n & 15;
  for (int kt = 0; kt < 17; ++kt)
    for (int hi = 0; hi < 2; ++hi) {
      v16h v;
      for (int e = 0; e < 16; ++e) {
        int c = kt * 32 + hi * 16 + e;
        float val = (c == 0) ? x0 : (c == 1) ? x1 : (c == 2) ? x2
                  : (c < 515) ? (float)fr[c - 3] : 0.f;
        v[e] = (_Float16)val;
      }
      reinterpret_cast<v16h*>(frag)[(nt * 17 + kt) * 32 + nl + hi * 16] = v;
    }
}

// ---------------------------------------------------------------- GEMM passes
// Pass 1: per-channel sum / sumsq of y = W*F (for BN mean/var).
// OB output tiles per wave; A register-cached; next B tile prefetched.
template <int KT, int OB>
__global__ void wmma_stats_kernel(const _Float16* __restrict__ WA,
                                  const _Float16* __restrict__ FB,
                                  int OTB, int NT, int strip,
                                  float* __restrict__ sums,
                                  float* __restrict__ sumsq) {
  int w    = (blockIdx.x * blockDim.x + threadIdx.x) >> 5;
  int lane = threadIdx.x & 31;
  int strips = (NT + strip - 1) / strip;
  if (w >= OTB * strips) return;                  // wave-uniform guard
  int ot0 = (w % OTB) * OB;
  int nt0 = (w / OTB) * strip;
  int nt1 = nt0 + strip; if (nt1 > NT) nt1 = NT;
  const v16h* A  = reinterpret_cast<const v16h*>(WA);
  const v16h* Bm = reinterpret_cast<const v16h*>(FB);
  v16h a[OB * KT];
  load_afrag<KT, OB>(A, ot0, lane, a);
  float s[OB][8] = {}, q[OB][8] = {};
  for (long nt = nt0; nt < nt1; ++nt) {
    __builtin_prefetch(Bm + ((nt + 1) * KT) * 32 + lane, 0, 1);
    v8f acc[OB];
    wmma_tiles<KT, OB>(a, Bm, nt, lane, acc);
#pragma unroll
    for (int ob = 0; ob < OB; ++ob)
      for (int r = 0; r < 8; ++r) {
        s[ob][r] += acc[ob][r];
        q[ob][r] += acc[ob][r] * acc[ob][r];
      }
  }
  for (int m = 1; m < 16; m <<= 1)
#pragma unroll
    for (int ob = 0; ob < OB; ++ob)
      for (int r = 0; r < 8; ++r) {
        s[ob][r] += __shfl_xor(s[ob][r], m);
        q[ob][r] += __shfl_xor(q[ob][r], m);
      }
  if ((lane & 15) == 0) {
    int hi = lane >> 4;
#pragma unroll
    for (int ob = 0; ob < OB; ++ob)
      for (int r = 0; r < 8; ++r) {
        int o = (ot0 + ob) * 16 + r + 8 * hi;
        atomicAdd(&sums[o],  s[ob][r]);
        atomicAdd(&sumsq[o], q[ob][r]);
      }
  }
}

__global__ void bn_finalize_kernel(const float* __restrict__ sums,
                                   const float* __restrict__ sumsq,
                                   const float* __restrict__ g,
                                   const float* __restrict__ bb,
                                   float cnt, int O,
                                   float* __restrict__ scale,
                                   float* __restrict__ bias) {
  int o = blockIdx.x * blockDim.x + threadIdx.x;
  if (o >= O) return;
  float mean = sums[o] / cnt;
  float var  = sumsq[o] / cnt - mean * mean;
  float sc   = g[o] * rsqrtf(var + 1e-5f);
  scale[o] = sc;
  bias[o]  = bb[o] - mean * sc;
}

// Pass 2a: recompute y, BN+ReLU, scatter into next layer's B-fragments
// (output channel o becomes next layer's K).
template <int KT, int OB>
__global__ void wmma_frag_kernel(const _Float16* __restrict__ WA,
                                 const _Float16* __restrict__ FB,
                                 int OTB, int NT,
                                 const float* __restrict__ scale,
                                 const float* __restrict__ bias,
                                 int KTn, _Float16* __restrict__ outF) {
  int w    = (blockIdx.x * blockDim.x + threadIdx.x) >> 5;
  int lane = threadIdx.x & 31;
  if (w >= NT * OTB) return;
  int  ot0 = (w % OTB) * OB;
  long nt  = w / OTB;
  const v16h* A  = reinterpret_cast<const v16h*>(WA);
  const v16h* Bm = reinterpret_cast<const v16h*>(FB);
  v16h a[OB * KT];
  load_afrag<KT, OB>(A, ot0, lane, a);
  v8f acc[OB];
  wmma_tiles<KT, OB>(a, Bm, nt, lane, acc);
  int hi = lane >> 4, nl = lane & 15;
#pragma unroll
  for (int ob = 0; ob < OB; ++ob)
    for (int r = 0; r < 8; ++r) {
      int o = (ot0 + ob) * 16 + r + 8 * hi;
      float v = fmaxf(acc[ob][r] * scale[o] + bias[o], 0.f);
      int  kk  = o & 31;
      long off = ((nt * KTn + (o >> 5)) * 32 + (nl | (((kk >> 4) & 1) << 4))) * 16 + (kk & 15);
      outF[off] = (_Float16)v;
    }
}

// Pass 2b: recompute y, BN+ReLU, max over the nsample group (GT tiles),
// write (group, O) as f16 pool buffer or f32 final output.
template <int KT, int OB>
__global__ void wmma_pool_kernel(const _Float16* __restrict__ WA,
                                 const _Float16* __restrict__ FB,
                                 int OTB, int G, int GT,
                                 const float* __restrict__ scale,
                                 const float* __restrict__ bias,
                                 _Float16* __restrict__ outH,
                                 float* __restrict__ outF, int O) {
  int w    = (blockIdx.x * blockDim.x + threadIdx.x) >> 5;
  int lane = threadIdx.x & 31;
  if (w >= G * OTB) return;
  int  ot0 = (w % OTB) * OB;
  long g   = w / OTB;
  int hi = lane >> 4;
  const v16h* A  = reinterpret_cast<const v16h*>(WA);
  const v16h* Bm = reinterpret_cast<const v16h*>(FB);
  v16h a[OB * KT];
  load_afrag<KT, OB>(A, ot0, lane, a);
  float sc[OB][8], bs[OB][8];
#pragma unroll
  for (int ob = 0; ob < OB; ++ob)
    for (int r = 0; r < 8; ++r) {
      int o = (ot0 + ob) * 16 + r + 8 * hi;
      sc[ob][r] = scale[o]; bs[ob][r] = bias[o];
    }
  float vmax[OB][8] = {};                          // ReLU >= 0
  for (int t = 0; t < GT; ++t) {
    long nt = g * GT + t;
    __builtin_prefetch(Bm + ((nt + 1) * KT) * 32 + lane, 0, 1);
    v8f acc[OB];
    wmma_tiles<KT, OB>(a, Bm, nt, lane, acc);
#pragma unroll
    for (int ob = 0; ob < OB; ++ob)
      for (int r = 0; r < 8; ++r)
        vmax[ob][r] = fmaxf(vmax[ob][r],
                            fmaxf(acc[ob][r] * sc[ob][r] + bs[ob][r], 0.f));
  }
  for (int m = 1; m < 16; m <<= 1)
#pragma unroll
    for (int ob = 0; ob < OB; ++ob)
      for (int r = 0; r < 8; ++r)
        vmax[ob][r] = fmaxf(vmax[ob][r], __shfl_xor(vmax[ob][r], m));
  if ((lane & 15) == 0) {
#pragma unroll
    for (int ob = 0; ob < OB; ++ob)
      for (int r = 0; r < 8; ++r) {
        int o = (ot0 + ob) * 16 + r + 8 * hi;
        if (outH) outH[g * O + o] = (_Float16)vmax[ob][r];
        else      outF[g * O + o] = vmax[ob][r];
      }
  }
}

// ---------------------------------------------------------------- host side
static inline int waves_grid(long waves) { return (int)((waves * 32 + 127) / 128); }

struct LayerCtx {
  float* stats; float* sums; float* sumsq; float* scl; float* bia;
  hipStream_t stream;
};

template <int KT, int OB>
static void run_layer(const LayerCtx& c,
                      const _Float16* WAp, const _Float16* FBp, int OT, int NT,
                      const float* gamma, const float* beta, float cnt, int O,
                      bool pool, int KTn, _Float16* outFrag,
                      int G, int GT, _Float16* outH, float* outFinal) {
  const int OTB = OT / OB;
  hipMemsetAsync(c.stats, 0, 1024 * sizeof(float), c.stream);
  int strips = (NT + 15) / 16;
  wmma_stats_kernel<KT, OB><<<waves_grid((long)OTB * strips), 128, 0, c.stream>>>(
      WAp, FBp, OTB, NT, 16, c.sums, c.sumsq);
  bn_finalize_kernel<<<(O + 63) / 64, 64, 0, c.stream>>>(
      c.sums, c.sumsq, gamma, beta, cnt, O, c.scl, c.bia);
  if (!pool) {
    wmma_frag_kernel<KT, OB><<<waves_grid((long)NT * OTB), 128, 0, c.stream>>>(
        WAp, FBp, OTB, NT, c.scl, c.bia, KTn, outFrag);
  } else {
    wmma_pool_kernel<KT, OB><<<waves_grid((long)G * OTB), 128, 0, c.stream>>>(
        WAp, FBp, OTB, G, GT, c.scl, c.bia, outH, outFinal, O);
  }
}

extern "C" void kernel_launch(void* const* d_in, const int* in_sizes, int n_in,
                              void* d_out, int out_size, void* d_ws, size_t ws_size,
                              hipStream_t stream) {
  (void)in_sizes; (void)n_in; (void)out_size; (void)ws_size;
  const float* pc  = (const float*)d_in[0];
  const float* w1a = (const float*)d_in[1];
  const float* g1a = (const float*)d_in[2];
  const float* b1a = (const float*)d_in[3];
  const float* w1b = (const float*)d_in[4];
  const float* g1b = (const float*)d_in[5];
  const float* b1b = (const float*)d_in[6];
  const float* w2a = (const float*)d_in[7];
  const float* g2a = (const float*)d_in[8];
  const float* b2a = (const float*)d_in[9];
  const float* w2b = (const float*)d_in[10];
  const float* g2b = (const float*)d_in[11];
  const float* b2b = (const float*)d_in[12];
  const float* w3  = (const float*)d_in[13];
  const float* g3  = (const float*)d_in[14];
  const float* b3  = (const float*)d_in[15];
  float* out = (float*)d_out;

  // workspace carve-out
  char*  ws  = (char*)d_ws;
  size_t off = 0;
  auto carve = [&](size_t bytes) {
    char* p = ws + off;
    off = (off + bytes + 255) & ~(size_t)255;
    return p;
  };
  int*      idx1  = (int*)carve(32 * 512 * 4);
  float*    xyz1  = (float*)carve(32 * 512 * 3 * 4);
  int*      idx2  = (int*)carve(32 * 128 * 4);
  float*    xyz2  = (float*)carve(32 * 128 * 3 * 4);
  _Float16* f1p   = (_Float16*)carve((size_t)32 * 512 * 128 * 2);
  _Float16* f2p   = (_Float16*)carve((size_t)32 * 128 * 512 * 2);
  _Float16* WA1a  = (_Float16*)carve(4  * 1  * 512 * 2);
  _Float16* WA1b  = (_Float16*)carve(8  * 2  * 512 * 2);
  _Float16* WA2a  = (_Float16*)carve(8  * 5  * 512 * 2);
  _Float16* WA2b  = (_Float16*)carve(32 * 4  * 512 * 2);
  _Float16* WA3   = (_Float16*)carve(32 * 17 * 512 * 2);
  float*    stats = (float*)carve(1024 * 4);        // sums[512] + sumsq[512]
  float*    scl   = (float*)carve(512 * 4);
  float*    bia   = (float*)carve(512 * 4);
  // big fragment regions (sequentially reused; all L2-resident at 192MB)
  _Float16* P1 = (_Float16*)carve((size_t)41943040 * 2); // frag1a / frag2a / frag3
  _Float16* P2 = (_Float16*)carve((size_t)50331648 * 2); // frag1b / frag2b
  _Float16* frag1a = P1, *frag2a = P1, *frag3 = P1;
  _Float16* frag1b = P2, *frag2b = P2;

  LayerCtx ctx{stats, stats, stats + 512, scl, bia, stream};

  // weight packing (A fragments)
  pack_weights_kernel<<<(4  * 1  * 32 + 63) / 64, 64, 0, stream>>>(w1a, WA1a,  64,   6,  4,  1);
  pack_weights_kernel<<<(8  * 2  * 32 + 63) / 64, 64, 0, stream>>>(w1b, WA1b, 128,  64,  8,  2);
  pack_weights_kernel<<<(8  * 5  * 32 + 63) / 64, 64, 0, stream>>>(w2a, WA2a, 128, 134,  8,  5);
  pack_weights_kernel<<<(32 * 4  * 32 + 63) / 64, 64, 0, stream>>>(w2b, WA2b, 512, 128, 32,  4);
  pack_weights_kernel<<<(32 * 17 * 32 + 63) / 64, 64, 0, stream>>>(w3,  WA3,  512, 515, 32, 17);

  // ---- Stage 1: FPS 2048->512, ball query r=0.23 K=48, MLP 6->64->128, maxpool
  fps_kernel<<<32, 256, 0, stream>>>(pc, 3 * NPTS, 1, NPTS, NPTS, 512, idx1, xyz1);
  group1_kernel<<<(32 * 512) / 2, 64, 0, stream>>>(pc, xyz1, frag1a);
  // layer 1a: 6->64 (KT=1, OB=2), NT=49152, write frags for 1b (KTn=2)
  run_layer<1, 2>(ctx, WA1a, frag1a, 4, 49152, g1a, b1a, 786432.f, 64,
                  false, 2, frag1b, 0, 0, nullptr, nullptr);
  // layer 1b: 64->128 (KT=2, OB=2), maxpool over k=48 (GT=3) -> f1p (b,s1,128)
  run_layer<2, 2>(ctx, WA1b, frag1b, 8, 49152, g1b, b1b, 786432.f, 128,
                  true, 0, nullptr, 32 * 512, 3, f1p, nullptr);

  // ---- Stage 2: FPS 512->128, ball query r=0.32 K=64, MLP 134->128->512, maxpool
  fps_kernel<<<32, 256, 0, stream>>>(xyz1, 512 * 3, 3, 1, 512, 128, idx2, xyz2);
  group2_kernel<<<(32 * 128) / 2, 64, 0, stream>>>(xyz1, xyz2, f1p, frag2a);
  // layer 2a: 134->128 (Kpad=160, KT=5, OB=2), NT=16384, write frags for 2b (KTn=4)
  run_layer<5, 2>(ctx, WA2a, frag2a, 8, 16384, g2a, b2a, 262144.f, 128,
                  false, 4, frag2b, 0, 0, nullptr, nullptr);
  // layer 2b: 128->512 (KT=4, OB=2), maxpool over k=64 (GT=4) -> f2p (b,s2,512)
  run_layer<4, 2>(ctx, WA2b, frag2b, 32, 16384, g2b, b2b, 262144.f, 512,
                  true, 0, nullptr, 32 * 128, 4, f2p, nullptr);

  // ---- Stage 3: 515->512 (Kpad=544, KT=17, OB=1) over (b, k=128) -> (32,512) f32
  gather3_kernel<<<(32 * 128 + 63) / 64, 64, 0, stream>>>(xyz2, f2p, frag3);
  run_layer<17, 1>(ctx, WA3, frag3, 32, 256, g3, b3, 4096.f, 512,
                   true, 0, nullptr, 32, 8, nullptr, out);
}
